// MAE_63144609186301
// MI455X (gfx1250) — matrix operand order repaired and checked by reading, not compiled, not run on hardware
//
#include <hip/hip_runtime.h>
#include <hip/hip_bf16.h>
#include <math.h>

// ---------------- problem constants (match reference) ----------------
#define BQ   64      // batch
#define PN   196     // patches
#define NT   197     // tokens (cls + patches)
#define ED   1024    // encoder dim
#define DDIM 512     // decoder dim
#define NHD  8       // heads
#define KEEPN 49     // PN * 0.25
#define EPSV 1e-5f

typedef __attribute__((ext_vector_type(16))) __bf16 v16bf;
typedef __attribute__((ext_vector_type(8)))  float  v8f;
typedef __attribute__((ext_vector_type(2)))  __bf16 v2bf;

#define FLAG_BIAS 1
#define FLAG_GELU 2
#define FLAG_RES  4
#define FLAG_TRB  8

#if defined(__has_builtin)
#if __has_builtin(__builtin_amdgcn_cvt_pk_bf16_f32)
#define HAVE_CVT_PK_BF16 1
#endif
#endif

// pack two f32 -> 2xbf16 dword with a single v_cvt_pk_bf16_f32
__device__ __forceinline__ unsigned pack2bf(float lo, float hi) {
#if defined(HAVE_CVT_PK_BF16)
  union { v2bf v; unsigned u; } p;
  p.v = __builtin_amdgcn_cvt_pk_bf16_f32(lo, hi);
  return p.u;
#elif defined(__gfx1250__)
  unsigned r;
  asm("v_cvt_pk_bf16_f32 %0, %1, %2" : "=v"(r) : "v"(lo), "v"(hi));
  return r;
#else
  union { __bf16 h[2]; unsigned u; } p;
  p.h[0] = (__bf16)lo;
  p.h[1] = (__bf16)hi;
  return p.u;
#endif
}

union Frag { v16bf v; uint4 u[2]; };

#define LDS_KW 20   // padded row stride in dwords (= 40 bf16) for 32-wide K tiles

// ---------------------------------------------------------------------
// Generic bf16-WMMA GEMM:  D = alpha * A(MxK) * B(KxN) [+bias][gelu][+Res]
// A fp32 row-major (lda); B fp32 row-major [K,N] (ldb) or, with TRB,
// stored [N,K] (ldb). Two-level batch z -> (b = z/nH, h = z%nH), per-operand
// strides. 256 threads = 8 waves; block tile 128x128, K-step 32.
// Software-pipelined: next tile's global loads overlap current WMMAs.
// Interior tiles take a clamp-free fast path.
// ---------------------------------------------------------------------
template <int TRB>
__global__ __launch_bounds__(256)
void k_gemm(const float* __restrict__ A, const float* __restrict__ Bm,
            const float* __restrict__ bias, const float* __restrict__ Res,
            float* __restrict__ D,
            int M, int N, int K, int lda, int ldb, int ldd, int ldr,
            long long sAb, long long sAh, long long sBb, long long sBh,
            long long sDb, long long sDh, long long sRb, long long sRh,
            int nH, int flags, float alpha)
{
  __shared__ unsigned As32[128 * LDS_KW];
  __shared__ unsigned Bs32[128 * LDS_KW];

  const int tid  = threadIdx.x;
  const int lane = tid & 31;
  const int wid  = tid >> 5;
  const int wm   = wid & 1;    // wave row (0..1) -> 64 rows each
  const int wn   = wid >> 1;   // wave col (0..3) -> 32 cols each

  const int z  = blockIdx.z;
  const int bb = z / nH, hh = z - bb * nH;
  const float* Ab = A  + (long long)bb * sAb + (long long)hh * sAh;
  const float* Bb = Bm + (long long)bb * sBb + (long long)hh * sBh;
  const long long offD = (long long)bb * sDb + (long long)hh * sDh;
  const long long offR = (long long)bb * sRb + (long long)hh * sRh;

  const int row0 = blockIdx.y * 128;
  const int col0 = blockIdx.x * 128;
  const bool inA = (row0 + 128 <= M);   // whole A tile rows valid
  const bool inB = (col0 + 128 <= N);   // whole B tile cols valid

  // per-thread staging coordinates (constant over K loop)
  // A (and B when TRB): p = t*256+tid -> element-pair (row = p>>4, kpair = (p&15)*2)
  const int rpA  = tid >> 4;          // + t*16 rows per iteration
  const int kpA  = (tid & 15) * 2;    // pair start within 32-wide K tile
  // B non-transposed: p -> (n = p&127, kpair = (p>>7)*2)
  const int nB0  = tid & 127;
  const int kpB0 = (tid >> 7) * 2;    // + t*4 per iteration

  float a0[8], a1[8], b0[8], b1[8];

  // ---- fetch tile at k0 into registers ----
  auto fetch = [&](int k0, bool fullK) {
    // ---------------- A ----------------
    if (inA && fullK) {
#pragma unroll
      for (int t = 0; t < 8; ++t) {
        const float* pa = Ab + (long long)(row0 + rpA + t * 16) * lda + (k0 + kpA);
        a0[t] = pa[0];
        a1[t] = pa[1];
      }
    } else {
#pragma unroll
      for (int t = 0; t < 8; ++t) {
        int gr = row0 + rpA + t * 16;
        int gk = k0 + kpA;
        int cr  = gr < M ? gr : M - 1;
        int ck0 = gk < K ? gk : K - 1;
        int ck1 = (gk + 1) < K ? (gk + 1) : K - 1;
        const float* pa = Ab + (long long)cr * lda;
        float va0 = pa[ck0];
        float va1 = pa[ck1];
        a0[t] = (gr < M && gk     < K) ? va0 : 0.0f;
        a1[t] = (gr < M && gk + 1 < K) ? va1 : 0.0f;
      }
    }
    // ---------------- B ----------------
    if (TRB) {
      if (inB && fullK) {
#pragma unroll
        for (int t = 0; t < 8; ++t) {
          const float* pb = Bb + (long long)(col0 + rpA + t * 16) * ldb + (k0 + kpA);
          b0[t] = pb[0];
          b1[t] = pb[1];
        }
      } else {
#pragma unroll
        for (int t = 0; t < 8; ++t) {
          int gc = col0 + rpA + t * 16;
          int gk = k0 + kpA;
          int cc  = gc < N ? gc : N - 1;
          int ck0 = gk < K ? gk : K - 1;
          int ck1 = (gk + 1) < K ? (gk + 1) : K - 1;
          const float* pb = Bb + (long long)cc * ldb;
          float vb0 = pb[ck0];
          float vb1 = pb[ck1];
          b0[t] = (gc < N && gk     < K) ? vb0 : 0.0f;
          b1[t] = (gc < N && gk + 1 < K) ? vb1 : 0.0f;
        }
      }
    } else {
      if (inB && fullK) {
#pragma unroll
        for (int t = 0; t < 8; ++t) {
          const float* pb = Bb + (long long)(k0 + kpB0 + t * 4) * ldb + (col0 + nB0);
          b0[t] = pb[0];
          b1[t] = pb[ldb];
        }
      } else {
#pragma unroll
        for (int t = 0; t < 8; ++t) {
          int gk = k0 + kpB0 + t * 4;
          int gc = col0 + nB0;
          int cc  = gc < N ? gc : N - 1;
          int ck0 = gk < K ? gk : K - 1;
          int ck1 = (gk + 1) < K ? (gk + 1) : K - 1;
          float vb0 = Bb[(long long)ck0 * ldb + cc];
          float vb1 = Bb[(long long)ck1 * ldb + cc];
          b0[t] = (gc < N && gk     < K) ? vb0 : 0.0f;
          b1[t] = (gc < N && gk + 1 < K) ? vb1 : 0.0f;
        }
      }
    }
  };

  // ---- store fetched registers into LDS as packed 2xbf16 dwords ----
  auto store_lds = [&]() {
#pragma unroll
    for (int t = 0; t < 8; ++t) {
      As32[(rpA + t * 16) * LDS_KW + (kpA >> 1)] = pack2bf(a0[t], a1[t]);
      if (TRB)
        Bs32[(rpA + t * 16) * LDS_KW + (kpA >> 1)] = pack2bf(b0[t], b1[t]);
      else
        Bs32[nB0 * LDS_KW + ((kpB0 + t * 4) >> 1)] = pack2bf(b0[t], b1[t]);
    }
  };

  v8f acc[4][2];
#pragma unroll
  for (int i = 0; i < 4; ++i)
#pragma unroll
    for (int j = 0; j < 2; ++j)
#pragma unroll
      for (int r = 0; r < 8; ++r) acc[i][j][r] = 0.0f;

  const int mrow = lane & 15;
  const int hf   = lane >> 4;
  const unsigned short* Asu = (const unsigned short*)As32;
  const unsigned short* Bsu = (const unsigned short*)Bs32;

  fetch(0, 32 <= K);
  for (int k0 = 0; k0 < K; k0 += 32) {
    __syncthreads();          // LDS consumers of previous tile done
    store_lds();
    __syncthreads();
    if (k0 + 32 < K) fetch(k0 + 32, k0 + 64 <= K);  // overlap with WMMAs

    // ---- fragment loads per ISA layouts ----
    // A 16-bit 16x32: lane half h: elems 0..7 = K 8h.., elems 8..15 = K 16+8h..
    Frag af[4], bfr[2];
#pragma unroll
    for (int mi = 0; mi < 4; ++mi) {
      int m = wm * 64 + mi * 16 + mrow;
      const unsigned short* p = &Asu[m * (2 * LDS_KW) + 8 * hf];
      af[mi].u[0] = *(const uint4*)(p);
      af[mi].u[1] = *(const uint4*)(p + 16);
    }
    // B 16-bit 32x16: lane n = lane&15, elems 0..7 = K 8h.., elems 8..15 = K 16+8h..
#pragma unroll
    for (int ni = 0; ni < 2; ++ni) {
      int n = wn * 32 + ni * 16 + mrow;
      const unsigned short* p = &Bsu[n * (2 * LDS_KW) + 8 * hf];
      bfr[ni].u[0] = *(const uint4*)(p);
      bfr[ni].u[1] = *(const uint4*)(p + 16);
    }
#pragma unroll
    for (int mi = 0; mi < 4; ++mi)
#pragma unroll
      for (int ni = 0; ni < 2; ++ni)
        acc[mi][ni] = __builtin_amdgcn_wmma_f32_16x16x32_bf16(
            false, af[mi].v, false, bfr[ni].v, (short)0, acc[mi][ni],
            false, false);
  }

  // ---- epilogue: C/D layout VGPR r -> M = r + 8*(lane/16), N = lane&15 ----
#pragma unroll
  for (int mi = 0; mi < 4; ++mi)
#pragma unroll
    for (int ni = 0; ni < 2; ++ni)
#pragma unroll
      for (int r = 0; r < 8; ++r) {
        int row = row0 + wm * 64 + mi * 16 + r + 8 * hf;
        int col = col0 + wn * 32 + ni * 16 + mrow;
        if (row < M && col < N) {
          float v = acc[mi][ni][r] * alpha;
          if (flags & FLAG_BIAS) v += bias[col];
          if (flags & FLAG_GELU) v = 0.5f * v * (1.0f + erff(v * 0.70710678118654752f));
          if (flags & FLAG_RES)  v += Res[offR + (long long)row * ldr + col];
          D[offD + (long long)row * ldd + col] = v;
        }
      }
}

// ---------------------------------------------------------------------
// Support kernels
// ---------------------------------------------------------------------
__global__ void k_zero(float* p) { if (threadIdx.x == 0 && blockIdx.x == 0) p[0] = 0.0f; }

// ids_restore = rank of noise[i] (stable argsort-of-argsort); mask = rank >= KEEP
__global__ __launch_bounds__(256)
void k_rank(const float* __restrict__ noise, int* __restrict__ ids, float* __restrict__ mask)
{
  __shared__ float nz[PN];
  int b = blockIdx.x;
  for (int i = threadIdx.x; i < PN; i += blockDim.x) nz[i] = noise[b * PN + i];
  __syncthreads();
  for (int i = threadIdx.x; i < PN; i += blockDim.x) {
    float vi = nz[i];
    int r = 0;
    for (int j = 0; j < PN; ++j) {
      float vj = nz[j];
      r += (vj < vi) || (vj == vi && j < i);
    }
    ids[b * PN + i]  = r;
    mask[b * PN + i] = (r >= KEEPN) ? 1.0f : 0.0f;
  }
}

// patches[b, p, c*256+u*16+v] = imgs[b, c, i*16+u, j*16+v], p = i*14+j
__global__ __launch_bounds__(256)
void k_patches(const float* __restrict__ imgs, float* __restrict__ P)
{
  long long idx = (long long)blockIdx.x * 256 + threadIdx.x;
  long long total = (long long)BQ * PN * 768;
  if (idx >= total) return;
  int d = (int)(idx % 768);
  int p = (int)((idx / 768) % PN);
  int b = (int)(idx / (768LL * PN));
  int c = d >> 8, rem = d & 255, u = rem >> 4, v = rem & 15;
  int i = p / 14, j = p % 14;
  P[idx] = imgs[(((long long)b * 3 + c) * 224 + (i * 16 + u)) * 224 + (j * 16 + v)];
}

// x = BN(x) + POS_E[1+p]  (pos computed analytically: 2D sincos)
__global__ __launch_bounds__(256)
void k_embed_post(float* __restrict__ X, const float* __restrict__ g,
                  const float* __restrict__ bb, const float* __restrict__ mean,
                  const float* __restrict__ var)
{
  long long idx = (long long)blockIdx.x * 256 + threadIdx.x;
  long long total = (long long)BQ * PN * ED;
  if (idx >= total) return;
  int d = (int)(idx % ED);
  int p = (int)((idx / ED) % PN);
  int quarter = d >> 8;       // ED/4 = 256
  int dd_ = d & 255;
  float omega = __powf(10000.0f, -(float)dd_ * (1.0f / 256.0f));
  int pi = p / 14, pj = p % 14;
  float mv = (quarter < 2) ? (float)pj : (float)pi;
  float pos = ((quarter & 1) == 0) ? __sinf(mv * omega) : __cosf(mv * omega);
  float v = X[idx];
  v = (v - mean[d]) * rsqrtf(var[d] + EPSV) * g[d] + bb[d] + pos;
  X[idx] = v;
}

// tok[b,0,:] = cls ; tok[b,1+t,:] = x[b, ids[b,t], :]
__global__ __launch_bounds__(256)
void k_build_tok(const float* __restrict__ X, const float* __restrict__ cls,
                 const int* __restrict__ ids, float* __restrict__ T)
{
  long long idx = (long long)blockIdx.x * 256 + threadIdx.x;
  long long total = (long long)BQ * NT * ED;
  if (idx >= total) return;
  int d = (int)(idx % ED);
  int n = (int)((idx / ED) % NT);
  int b = (int)(idx / ((long long)ED * NT));
  float v;
  if (n == 0) v = cls[d];
  else {
    int src = ids[b * PN + (n - 1)];
    v = X[((long long)b * PN + src) * ED + d];
  }
  T[idx] = v;
}

// allp[b,0,:] = xd[b,0,:]; allp[b,1+t,:] = xd[b,1+ids[b,t],:] ; + POS_D[n]
__global__ __launch_bounds__(256)
void k_build_dec(const float* __restrict__ XD, const int* __restrict__ ids,
                 float* __restrict__ T)
{
  long long idx = (long long)blockIdx.x * 256 + threadIdx.x;
  long long total = (long long)BQ * NT * DDIM;
  if (idx >= total) return;
  int d = (int)(idx % DDIM);
  int n = (int)((idx / DDIM) % NT);
  int b = (int)(idx / ((long long)DDIM * NT));
  float v;
  float pos = 0.0f;
  if (n == 0) {
    v = XD[((long long)b * NT) * DDIM + d];
  } else {
    int r = ids[b * PN + (n - 1)];
    v = XD[((long long)b * NT + 1 + r) * DDIM + d];
    int p = n - 1;
    int quarter = d >> 7;     // DDIM/4 = 128
    int dd_ = d & 127;
    float omega = __powf(10000.0f, -(float)dd_ * (1.0f / 256.0f)); // /256 like ref
    int pi = p / 14, pj = p % 14;
    float mv = (quarter < 2) ? (float)pj : (float)pi;
    pos = ((quarter & 1) == 0) ? __sinf(mv * omega) : __cosf(mv * omega);
  }
  T[idx] = v + pos;
}

__global__ __launch_bounds__(256)
void k_layernorm(const float* __restrict__ X, float* __restrict__ Y,
                 const float* __restrict__ g, const float* __restrict__ b, int C)
{
  __shared__ float s1[256], s2[256];
  long long row = blockIdx.x;
  const float* x = X + row * C;
  float a = 0.0f, q = 0.0f;
  for (int c = threadIdx.x; c < C; c += 256) { float v = x[c]; a += v; q += v * v; }
  s1[threadIdx.x] = a; s2[threadIdx.x] = q; __syncthreads();
  for (int s = 128; s > 0; s >>= 1) {
    if ((int)threadIdx.x < s) { s1[threadIdx.x] += s1[threadIdx.x + s]; s2[threadIdx.x] += s2[threadIdx.x + s]; }
    __syncthreads();
  }
  float mu  = s1[0] / (float)C;
  float var = s2[0] / (float)C - mu * mu;
  float inv = rsqrtf(var + EPSV);
  for (int c = threadIdx.x; c < C; c += 256)
    Y[row * C + c] = (x[c] - mu) * inv * g[c] + b[c];
}

__global__ __launch_bounds__(128)
void k_softmax(float* __restrict__ X, int len)
{
  __shared__ float red[128];
  long long row = blockIdx.x;
  float* x = X + row * len;
  float mx = -1e30f;
  for (int c = threadIdx.x; c < len; c += 128) mx = fmaxf(mx, x[c]);
  red[threadIdx.x] = mx; __syncthreads();
  for (int s = 64; s > 0; s >>= 1) {
    if ((int)threadIdx.x < s) red[threadIdx.x] = fmaxf(red[threadIdx.x], red[threadIdx.x + s]);
    __syncthreads();
  }
  mx = red[0]; __syncthreads();
  float sm = 0.0f;
  for (int c = threadIdx.x; c < len; c += 128) { float e = __expf(x[c] - mx); x[c] = e; sm += e; }
  red[threadIdx.x] = sm; __syncthreads();
  for (int s = 64; s > 0; s >>= 1) {
    if ((int)threadIdx.x < s) red[threadIdx.x] += red[threadIdx.x + s];
    __syncthreads();
  }
  float inv = 1.0f / red[0];
  for (int c = threadIdx.x; c < len; c += 128) x[c] *= inv;
}

// writes sliced allp output and accumulates masked MSE into loss[0]
__global__ __launch_bounds__(256)
void k_loss_out(const float* __restrict__ lin, const float* __restrict__ imgs,
                const float* __restrict__ mask, float* __restrict__ out_allp,
                float* __restrict__ loss, long long total)
{
  __shared__ float red[256];
  float acc = 0.0f;
  for (long long idx = (long long)blockIdx.x * 256 + threadIdx.x; idx < total;
       idx += (long long)gridDim.x * 256) {
    int j = (int)(idx % 768);
    long long r = idx / 768;
    int t = (int)(r % PN);
    int b = (int)(r / PN);
    float v = lin[((long long)b * NT + 1 + t) * 768 + j];
    out_allp[idx] = v;
    int c = j % 3, uv = j / 3, u = uv >> 4, vv = uv & 15;
    int pi = t / 14, pj = t % 14;
    float y = imgs[(((long long)b * 3 + c) * 224 + (pi * 16 + u)) * 224 + (pj * 16 + vv)];
    float m = mask[b * PN + t];
    float d = (v - y) * m;
    acc += d * d;
  }
  red[threadIdx.x] = acc; __syncthreads();
  for (int s = 128; s > 0; s >>= 1) {
    if ((int)threadIdx.x < s) red[threadIdx.x] += red[threadIdx.x + s];
    __syncthreads();
  }
  if (threadIdx.x == 0) atomicAdd(loss, red[0] / (float)total);
}

// ---------------------------------------------------------------------
// host-side helpers
// ---------------------------------------------------------------------
static inline void gemm(hipStream_t st,
    const float* A, const float* B, const float* bias, const float* Res, float* D,
    int M, int N, int K, int lda, int ldb, int ldd, int ldr,
    long long sAb, long long sAh, long long sBb, long long sBh,
    long long sDb, long long sDh, long long sRb, long long sRh,
    int nB, int nH, int flags, float alpha)
{
  dim3 g((N + 127) / 128, (M + 127) / 128, nB * nH);
  if (flags & FLAG_TRB)
    k_gemm<1><<<g, 256, 0, st>>>(A, B, bias, Res, D, M, N, K, lda, ldb, ldd, ldr,
                                 sAb, sAh, sBb, sBh, sDb, sDh, sRb, sRh,
                                 nH, flags, alpha);
  else
    k_gemm<0><<<g, 256, 0, st>>>(A, B, bias, Res, D, M, N, K, lda, ldb, ldd, ldr,
                                 sAb, sAh, sBb, sBh, sDb, sDh, sRb, sRh,
                                 nH, flags, alpha);
}

static void tblock(hipStream_t st, float* W0, float* W1, float* W2, float* W3,
                   float* W4, float* W5, float* W6,
                   const float* lng, const float* lnb,
                   const float* qkvw, const float* qkvb,
                   const float* w0w, const float* w0b,
                   const float* m1w, const float* m1b,
                   const float* m2w, const float* m2b,
                   int C, int MLP)
{
  const int rows = BQ * NT;
  const int dh = C / NHD;
  // LN1
  k_layernorm<<<rows, 256, 0, st>>>(W0, W1, lng, lnb, C);
  // QKV
  gemm(st, W1, qkvw, qkvb, nullptr, W2, rows, 3 * C, C, C, 3 * C, 3 * C, 0,
       0, 0, 0, 0, 0, 0, 0, 0, 1, 1, FLAG_BIAS, 1.0f);
  // S = (Q K^T) * sqrt(dh)   (matches softmax(QK^T / dh^-0.5))
  gemm(st, W2, W2 + C, nullptr, nullptr, W3, NT, NT, dh, 3 * C, 3 * C, NT, 0,
       (long long)NT * 3 * C, dh, (long long)NT * 3 * C, dh,
       (long long)NHD * NT * NT, (long long)NT * NT, 0, 0,
       BQ, NHD, FLAG_TRB, sqrtf((float)dh));
  k_softmax<<<BQ * NHD * NT, 128, 0, st>>>(W3, NT);
  // O = S V  (written head-interleaved into [B*NT, C])
  gemm(st, W3, W2 + 2 * C, nullptr, nullptr, W4, NT, dh, NT, NT, 3 * C, C, 0,
       (long long)NHD * NT * NT, (long long)NT * NT, (long long)NT * 3 * C, dh,
       (long long)NT * C, dh, 0, 0, BQ, NHD, 0, 1.0f);
  // proj + residual
  gemm(st, W4, w0w, w0b, W0, W5, rows, C, C, C, C, C, C,
       0, 0, 0, 0, 0, 0, 0, 0, 1, 1, FLAG_BIAS | FLAG_RES, 1.0f);
  // LN2
  k_layernorm<<<rows, 256, 0, st>>>(W5, W1, lng, lnb, C);
  // MLP
  gemm(st, W1, m1w, m1b, nullptr, W6, rows, MLP, C, C, MLP, MLP, 0,
       0, 0, 0, 0, 0, 0, 0, 0, 1, 1, FLAG_BIAS | FLAG_GELU, 1.0f);
  gemm(st, W6, m2w, m2b, W5, W0, rows, C, MLP, MLP, C, C, C,
       0, 0, 0, 0, 0, 0, 0, 0, 1, 1, FLAG_BIAS | FLAG_RES, 1.0f);
}

extern "C" void kernel_launch(void* const* d_in, const int* in_sizes, int n_in,
                              void* d_out, int out_size, void* d_ws, size_t ws_size,
                              hipStream_t stream)
{
  (void)in_sizes; (void)n_in; (void)out_size; (void)ws_size;
  const float* imgs      = (const float*)d_in[0];
  const float* noise     = (const float*)d_in[1];
  const float* conv_w    = (const float*)d_in[2];
  const float* conv_b    = (const float*)d_in[3];
  const float* bn_g      = (const float*)d_in[4];
  const float* bn_b      = (const float*)d_in[5];
  const float* bn_mean   = (const float*)d_in[6];
  const float* bn_var    = (const float*)d_in[7];
  const float* cls_token = (const float*)d_in[8];
  const float* qkv_w_e   = (const float*)d_in[9];
  const float* qkv_b_e   = (const float*)d_in[10];
  const float* w0_w_e    = (const float*)d_in[11];
  const float* w0_b_e    = (const float*)d_in[12];
  const float* m1w_e     = (const float*)d_in[13];
  const float* m1b_e     = (const float*)d_in[14];
  const float* m2w_e     = (const float*)d_in[15];
  const float* m2b_e     = (const float*)d_in[16];
  const float* ln_g_e    = (const float*)d_in[17];
  const float* ln_b_e    = (const float*)d_in[18];
  const float* de_w      = (const float*)d_in[19];
  const float* de_b      = (const float*)d_in[20];
  // d_in[21] = mask_patch : never selected by the gather in this reference
  const float* qkv_w_d   = (const float*)d_in[22];
  const float* qkv_b_d   = (const float*)d_in[23];
  const float* w0_w_d    = (const float*)d_in[24];
  const float* w0_b_d    = (const float*)d_in[25];
  const float* m1w_d     = (const float*)d_in[26];
  const float* m1b_d     = (const float*)d_in[27];
  const float* m2w_d     = (const float*)d_in[28];
  const float* m2b_d     = (const float*)d_in[29];
  const float* ln_g_d    = (const float*)d_in[30];
  const float* ln_b_d    = (const float*)d_in[31];
  const float* lin_w     = (const float*)d_in[32];
  const float* lin_b     = (const float*)d_in[33];

  float* out = (float*)d_out;

  // ---- workspace layout (floats) ----
  int*   ids   = (int*)d_ws;                         // B*PN ints
  float* maskf = (float*)d_ws + BQ * PN;             // B*PN
  float* base  = (float*)d_ws + 2 * BQ * PN;
  const long long n0 = (long long)BQ * NT * ED;      // 12,910,592
  float* W0 = base;                                  // tokens
  float* W1 = W0 + n0;                               // LN out
  float* W2 = W1 + n0;                               // QKV (B*NT*3C)
  float* W3 = W2 + (long long)BQ * NT * 3 * ED;      // attention scores
  float* W4 = W3 + (long long)BQ * NHD * NT * NT;    // attn out / xd / head out
  float* W5 = W4 + n0;                               // z / x_embed
  float* W6 = W5 + n0;                               // mlp hidden / patches

  k_zero<<<1, 1, 0, stream>>>(out);
  k_rank<<<BQ, 256, 0, stream>>>(noise, ids, maskf);

  // ---- patch embed ----
  {
    long long n = (long long)BQ * PN * 768;
    k_patches<<<(unsigned)((n + 255) / 256), 256, 0, stream>>>(imgs, W6);
  }
  // patches [B*PN,768] x conv_w^T(1024x768) -> W5 [B*PN, 1024]
  gemm(stream, W6, conv_w, conv_b, nullptr, W5, BQ * PN, ED, 768, 768, 768, ED, 0,
       0, 0, 0, 0, 0, 0, 0, 0, 1, 1, FLAG_BIAS | FLAG_TRB, 1.0f);
  {
    long long n = (long long)BQ * PN * ED;
    k_embed_post<<<(unsigned)((n + 255) / 256), 256, 0, stream>>>(W5, bn_g, bn_b, bn_mean, bn_var);
  }
  {
    long long n = (long long)BQ * NT * ED;
    k_build_tok<<<(unsigned)((n + 255) / 256), 256, 0, stream>>>(W5, cls_token, ids, W0);
  }

  // ---- encoder ----
  for (int l = 0; l < 12; ++l)
    tblock(stream, W0, W1, W2, W3, W4, W5, W6,
           ln_g_e, ln_b_e, qkv_w_e, qkv_b_e, w0_w_e, w0_b_e,
           m1w_e, m1b_e, m2w_e, m2b_e, ED, 1500);

  // x_mask output
  hipMemcpyAsync(out + 1, W0, n0 * sizeof(float), hipMemcpyDeviceToDevice, stream);

  // ---- decoder embed + token rebuild ----
  gemm(stream, W0, de_w, de_b, nullptr, W4, BQ * NT, DDIM, ED, ED, DDIM, DDIM, 0,
       0, 0, 0, 0, 0, 0, 0, 0, 1, 1, FLAG_BIAS, 1.0f);
  {
    long long n = (long long)BQ * NT * DDIM;
    k_build_dec<<<(unsigned)((n + 255) / 256), 256, 0, stream>>>(W4, ids, W0);
  }

  // ---- decoder ----
  for (int l = 0; l < 12; ++l)
    tblock(stream, W0, W1, W2, W3, W4, W5, W6,
           ln_g_d, ln_b_d, qkv_w_d, qkv_b_d, w0_w_d, w0_b_d,
           m1w_d, m1b_d, m2w_d, m2b_d, DDIM, 1500);

  // ---- prediction head ----
  gemm(stream, W0, lin_w, lin_b, nullptr, W4, BQ * NT, 768, DDIM, DDIM, 768, 768, 0,
       0, 0, 0, 0, 0, 0, 0, 0, 1, 1, FLAG_BIAS, 1.0f);

  // ---- loss + sliced allp output ----
  k_loss_out<<<2048, 256, 0, stream>>>(W4, imgs, maskf,
                                       out + 1 + n0, out,
                                       (long long)BQ * PN * 768);
}
